// MoELayer_83837761618649
// MI455X (gfx1250) — matrix operand order repaired and checked by reading, not compiled, not run on hardware
//
#include <hip/hip_runtime.h>
#include <hip/hip_bf16.h>

// ---------------------------------------------------------------------------
// MoE top-1 FFN for MI455X (gfx1250, wave32, WMMA bf16 16x16x32)
// B=2 S=2048 D=1024 H=4096 E=8  -> N=4096 tokens
// ---------------------------------------------------------------------------

#define N_TOK 4096
#define D_ 1024
#define H_ 4096
#define E_ 8
#define NT 16                 // tokens per tile (= WMMA M)
#define LDX (D_ + 8)          // LDS stride (bf16 elems): 2064B/row = 129*16B
#define LDH (H_ + 8)          // LDS stride (bf16 elems): 8208B/row = 513*16B

typedef __attribute__((ext_vector_type(16))) __bf16 v16bf;
typedef __attribute__((ext_vector_type(8)))  float  v8f;

// Load a WMMA A-fragment (16x32 bf16, ISA wave32 layout): two contiguous
// 16-byte chunks per lane -> two ds_load_b128, no element shuffles.
__device__ __forceinline__ v16bf load_a_frag(const __bf16* pr) {
    union { v16bf v; uint4 q[2]; } u;
    u.q[0] = *(const uint4*)(pr);
    u.q[1] = *(const uint4*)(pr + 16);
    return u.v;
}

// ---------------------------------------------------------------------------
// Kernel 0: zero expert counters + l_aux output scalar
// ---------------------------------------------------------------------------
__global__ void moe_init_kernel(int* counts, float* laux) {
    if (threadIdx.x < E_) counts[threadIdx.x] = 0;
    if (threadIdx.x == 0) *laux = 0.0f;
}

// ---------------------------------------------------------------------------
// Kernel 1: gating. scores = x @ gate_w + gate_b ; first-occurrence argmax.
// Builds per-expert token lists via atomics.
// ---------------------------------------------------------------------------
__global__ __launch_bounds__(256)
void moe_gate_kernel(const float* __restrict__ x,
                     const float* __restrict__ gw,
                     const float* __restrict__ gb,
                     int* __restrict__ counts,
                     int* __restrict__ lists) {
    int tkn = blockIdx.x * blockDim.x + threadIdx.x;
    if (tkn >= N_TOK) return;
    const float* xr = x + (size_t)tkn * D_;
    float s[E_];
#pragma unroll
    for (int e = 0; e < E_; ++e) s[e] = gb[e];
    for (int d = 0; d < D_; d += 4) {
        float4 xv = *(const float4*)(xr + d);
        const float* g0 = gw + (size_t)d * E_;
#pragma unroll
        for (int e = 0; e < E_; ++e) {
            s[e] += xv.x * g0[e] + xv.y * g0[E_ + e] +
                    xv.z * g0[2 * E_ + e] + xv.w * g0[3 * E_ + e];
        }
    }
    int best = 0; float bv = s[0];
#pragma unroll
    for (int e = 1; e < E_; ++e)
        if (s[e] > bv) { bv = s[e]; best = e; }   // strict > == first argmax
    int pos = atomicAdd(&counts[best], 1);
    lists[best * N_TOK + pos] = tkn;
}

// ---------------------------------------------------------------------------
// Kernel 2: routed FFN. One block = one (expert, 16-token tile).
// Phase 1: Hs = gelu(X @ W1[e] + b1[e])   (bf16 staged in LDS, ~128KB)
// Phase 2: out = Hs @ W2[e] + b2[e]       (masked scatter, fp32)
// Dynamic LDS ~160KB -- needs CDNA5's 320KB-per-WGP LDS.
// ---------------------------------------------------------------------------
extern __shared__ __bf16 smem[];   // [NT*LDX] X tile | [NT*LDH] H tile

__global__ __launch_bounds__(256)
void moe_ffn_kernel(const float* __restrict__ x,
                    const float* __restrict__ w1,
                    const float* __restrict__ b1,
                    const float* __restrict__ w2,
                    const float* __restrict__ b2,
                    float* __restrict__ out,
                    const int* __restrict__ counts,
                    const int* __restrict__ lists) {
    const int tilesPerE = N_TOK / NT;                 // 256
    const int e    = blockIdx.x / tilesPerE;
    const int t    = blockIdx.x % tilesPerE;
    const int nt   = counts[e];
    const int base = t * NT;
    if (base >= nt) return;                           // block-uniform exit

    __shared__ int tok[NT];
    const int tid = threadIdx.x;
    if (tid < NT) {
        int idx = base + tid;
        tok[tid] = lists[e * N_TOK + (idx < nt ? idx : base)];
    }
    __syncthreads();

    __bf16* Xs = smem;
    __bf16* Hs = smem + NT * LDX;

    // ---- stage X tile: 16 gathered rows, fp32 -> bf16 (hw cvt), float4 ----
    for (int i = tid; i < NT * (D_ / 4); i += 256) {
        int r  = i / (D_ / 4);
        int c4 = i % (D_ / 4);
        float4 v = ((const float4*)(x + (size_t)tok[r] * D_))[c4];
        int c = c4 * 4;
        Xs[r * LDX + c + 0] = (__bf16)v.x;
        Xs[r * LDX + c + 1] = (__bf16)v.y;
        Xs[r * LDX + c + 2] = (__bf16)v.z;
        Xs[r * LDX + c + 3] = (__bf16)v.w;
    }
    __syncthreads();

    const int lane = tid & 31;
    const int wave = tid >> 5;
    const int hl   = lane >> 4;        // lane half (K interleave select)
    const int l16  = lane & 15;

    const float* W1 = w1 + (size_t)e * D_ * H_;
    const float* B1 = b1 + (size_t)e * H_;
    const float* W2 = w2 + (size_t)e * H_ * D_;
    const float* B2 = b2 + (size_t)e * D_;

    // ------------------- phase 1: H = gelu(X @ W1 + b1) -------------------
    for (int ct = wave * 32; ct < (wave + 1) * 32; ++ct) {
        const int col0 = ct * 16;
        const float bias = B1[col0 + l16];
        v8f c;
#pragma unroll
        for (int v = 0; v < 8; ++v) c[v] = bias;

        for (int k = 0; k < D_; k += 32) {
            // A fragment from LDS: two 16B chunks (ds_load_b128 x2)
            v16bf a = load_a_frag(&Xs[l16 * LDX + k + hl * 8]);

            // B fragment (32x16): lane half h holds K 8h..8h+7 / 16+8h..16+8h+7 of column
            v16bf b;
            const float* pw = W1 + (size_t)(k + hl * 8) * H_ + col0 + l16;
            __builtin_prefetch((const void*)(pw + (size_t)32 * H_), 0, 1);
#pragma unroll
            for (int j = 0; j < 16; ++j) b[j] = (__bf16)pw[(size_t)(j < 8 ? j : j + 8) * H_];

            c = __builtin_amdgcn_wmma_f32_16x16x32_bf16(
                    false, a, false, b, (short)0, c, false, false);
        }
        // exact gelu, stage bf16 into LDS H buffer
#pragma unroll
        for (int v = 0; v < 8; ++v) {
            int   m  = v + hl * 8;
            float hx = c[v];
            float g  = 0.5f * hx * (1.0f + erff(hx * 0.70710678118f));
            Hs[m * LDH + col0 + l16] = (__bf16)g;
        }
    }
    __syncthreads();

    // ------------------- phase 2: out = H @ W2 + b2 -----------------------
    for (int ct = wave * 8; ct < (wave + 1) * 8; ++ct) {
        const int col0 = ct * 16;
        const float bias = B2[col0 + l16];
        v8f c;
#pragma unroll
        for (int v = 0; v < 8; ++v) c[v] = bias;

        for (int k = 0; k < H_; k += 32) {
            v16bf a = load_a_frag(&Hs[l16 * LDH + k + hl * 8]);

            v16bf b;
            const float* pw = W2 + (size_t)(k + hl * 8) * D_ + col0 + l16;
            __builtin_prefetch((const void*)(pw + (size_t)32 * D_), 0, 1);
#pragma unroll
            for (int j = 0; j < 16; ++j) b[j] = (__bf16)pw[(size_t)(j < 8 ? j : j + 8) * D_];

            c = __builtin_amdgcn_wmma_f32_16x16x32_bf16(
                    false, a, false, b, (short)0, c, false, false);
        }
        // masked fp32 scatter (one writer per output row -> no zeroing needed)
#pragma unroll
        for (int v = 0; v < 8; ++v) {
            int m = v + hl * 8;
            if (base + m < nt) {
                out[(size_t)tok[m] * D_ + col0 + l16] = c[v];
            }
        }
    }
}

// ---------------------------------------------------------------------------
// Host launcher
// ---------------------------------------------------------------------------
extern "C" void kernel_launch(void* const* d_in, const int* in_sizes, int n_in,
                              void* d_out, int out_size, void* d_ws, size_t ws_size,
                              hipStream_t stream) {
    (void)in_sizes; (void)n_in; (void)ws_size;
    const float* x      = (const float*)d_in[0];
    const float* gate_w = (const float*)d_in[1];
    const float* gate_b = (const float*)d_in[2];
    const float* w1     = (const float*)d_in[3];
    const float* b1     = (const float*)d_in[4];
    const float* w2     = (const float*)d_in[5];
    const float* b2     = (const float*)d_in[6];

    float* out  = (float*)d_out;
    float* laux = out + (out_size - 1);   // tuple: [B*S*D floats][l_aux]

    int* counts = (int*)d_ws;                         // 8 ints
    int* lists  = (int*)((char*)d_ws + 128);          // E * N ints

    moe_init_kernel<<<1, 32, 0, stream>>>(counts, laux);

    moe_gate_kernel<<<N_TOK / 256, 256, 0, stream>>>(
        x, gate_w, gate_b, counts, lists);

    const size_t lds_bytes = (size_t)(NT * LDX + NT * LDH) * sizeof(__bf16); // ~160KB
    moe_ffn_kernel<<<E_ * (N_TOK / NT), 256, lds_bytes, stream>>>(
        x, w1, b1, w2, b2, out, counts, lists);
}
